// DeepSeekMLA_7894149890077
// MI455X (gfx1250) — matrix operand order repaired and checked by reading, not compiled, not run on hardware
//
#include <hip/hip_runtime.h>

// ---------------------------------------------------------------------------
// DeepSeek MLA prefill for MI455X (gfx1250, wave32, WMMA bf16 path)
// GEMMs: 128x128 block tile, double-buffered LDS staging via async
// global->LDS copies (ASYNCcnt), 8 waves each computing a 32x64 tile with
// v_wmma_f32_16x16x32_bf16. Attention: wave-level flash attention with
// ds_swizzle softmax reductions and LDS-based P relayout.
// ---------------------------------------------------------------------------

typedef __attribute__((ext_vector_type(16))) __bf16 v16bf;
typedef __attribute__((ext_vector_type(8)))  __bf16 v8bf;
typedef __attribute__((ext_vector_type(8)))  float  v8f;
typedef __attribute__((ext_vector_type(4)))  int    v4i;

union bfu { v16bf v; v8bf h[2]; };

#define WMMA_BF16(A, B, C) \
  __builtin_amdgcn_wmma_f32_16x16x32_bf16(false, (A), false, (B), (short)0, (C), false, false)

// float -> bf16 round-to-nearest-even (bit ops only)
__device__ __forceinline__ __bf16 f2bf(float f) {
  unsigned u = __builtin_bit_cast(unsigned, f);
  u += 0x7fffu + ((u >> 16) & 1u);
  unsigned short t = (unsigned short)(u >> 16);
  return __builtin_bit_cast(__bf16, t);
}

// ---------------------------------------------------------------------------
// Async global->LDS staging (CDNA5 GLOBAL_LOAD_ASYNC_TO_LDS_B128, ASYNCcnt).
// Confirmed present on this toolchain; parameter types are <4 x i32>* in
// addrspace(1)/(3) per the round-2 diagnostic.
// ---------------------------------------------------------------------------
#if __has_builtin(__builtin_amdgcn_global_load_async_to_lds_b128)
#define HAS_ASYNC_LDS 1
#else
#define HAS_ASYNC_LDS 0
#endif

typedef __attribute__((address_space(1))) v4i gv4i;
typedef __attribute__((address_space(3))) v4i lv4i;

__device__ __forceinline__ void stage_b128(const __bf16* g, __bf16* l) {
#if HAS_ASYNC_LDS
  __builtin_amdgcn_global_load_async_to_lds_b128((gv4i*)(g), (lv4i*)(l), 0, 0);
#else
  *(v8bf*)l = *(const v8bf*)g;
#endif
}

#if __has_builtin(__builtin_amdgcn_s_wait_asynccnt)
#define WAIT_ASYNC(n) __builtin_amdgcn_s_wait_asynccnt(n)
#elif HAS_ASYNC_LDS
#define WAIT_ASYNC(n) asm volatile("s_wait_asynccnt %0" ::"n"(n) : "memory")
#else
#define WAIT_ASYNC(n)
#endif

// ds_swizzle XOR-mode lane reductions within each 16-lane half of the wave
#define SWZ(v, imm) \
  __builtin_bit_cast(float, __builtin_amdgcn_ds_swizzle(__builtin_bit_cast(int, (v)), (imm)))

__device__ __forceinline__ float half_max16(float v) {
  v = fmaxf(v, SWZ(v, 0x041f));
  v = fmaxf(v, SWZ(v, 0x081f));
  v = fmaxf(v, SWZ(v, 0x101f));
  v = fmaxf(v, SWZ(v, 0x201f));
  return v;
}
__device__ __forceinline__ float half_sum16(float v) {
  v += SWZ(v, 0x041f);
  v += SWZ(v, 0x081f);
  v += SWZ(v, 0x101f);
  v += SWZ(v, 0x201f);
  return v;
}

// ---------------------------------------------------------------------------
// elementwise: fp32 -> bf16, and bf16 zero fill (for N-padding)
// ---------------------------------------------------------------------------
__global__ void cvt_f32_bf16(const float* __restrict__ in, __bf16* __restrict__ out, long n) {
  long i = (long)blockIdx.x * blockDim.x + threadIdx.x;
  long st = (long)gridDim.x * blockDim.x;
  for (; i < n; i += st) out[i] = f2bf(in[i]);
}
__global__ void zero_bf16(__bf16* __restrict__ p, long n) {
  long i = (long)blockIdx.x * blockDim.x + threadIdx.x;
  long st = (long)gridDim.x * blockDim.x;
  for (; i < n; i += st) p[i] = f2bf(0.f);
}

// ---------------------------------------------------------------------------
// GEMM  C[M,N](f32) = A[M,K](bf16) @ B[N,K](bf16)^T
// block = 256 thr = 8 waves, block tile 128x128, K-step 32.
// A/B 128x32 tiles double-buffered in LDS (row stride padded to 40 elems =
// 80B: keeps every 16B chunk aligned and spreads LDS banks).
// Waves: 4(M) x 2(N); each wave computes 32x64 (2x4 accum tiles).
// Requires M%128==0, N%128==0, K%32==0.
// ---------------------------------------------------------------------------
#define LDS_STRIDE 40

__global__ __launch_bounds__(256) void gemm_bf16(const __bf16* __restrict__ A,
                                                 const __bf16* __restrict__ B,
                                                 float* __restrict__ C,
                                                 int M, int N, int K) {
  __shared__ __bf16 As[2][128 * LDS_STRIDE];
  __shared__ __bf16 Bs[2][128 * LDS_STRIDE];
  const int tid = threadIdx.x;
  const int wave = tid >> 5, lane = tid & 31;
  const int l15 = lane & 15, hs = lane >> 4;
  const int wm = wave & 3, wn = wave >> 2;
  const int mb = blockIdx.y * 128;
  const int nb = blockIdx.x * 128;

  // staging: 128 rows x 4 chunks(16B) per tile = 512 chunks; 2 per thread/tile
  const int r0 = tid >> 2;  // 0..63
  const int c0 = tid & 3;   // chunk 0..3

  auto issue = [&](int buf, int k0) {
#pragma unroll
    for (int i = 0; i < 2; ++i) {
      const int r = r0 + i * 64;
      stage_b128(A + (size_t)(mb + r) * K + k0 + c0 * 8, &As[buf][r * LDS_STRIDE + c0 * 8]);
      stage_b128(B + (size_t)(nb + r) * K + k0 + c0 * 8, &Bs[buf][r * LDS_STRIDE + c0 * 8]);
    }
  };

  const v8f vzero = {0.f, 0.f, 0.f, 0.f, 0.f, 0.f, 0.f, 0.f};
  v8f acc[2][4];
#pragma unroll
  for (int s = 0; s < 2; ++s)
#pragma unroll
    for (int t = 0; t < 4; ++t) acc[s][t] = vzero;

  auto compute = [&](int buf) {
    v16bf af[2];
#pragma unroll
    for (int s = 0; s < 2; ++s) {
      bfu u;
      const __bf16* ap = &As[buf][(wm * 32 + s * 16 + l15) * LDS_STRIDE + hs * 8];
      u.h[0] = *(const v8bf*)ap;
      u.h[1] = *(const v8bf*)(ap + 16);
      af[s] = u.v;
    }
#pragma unroll
    for (int t = 0; t < 4; ++t) {
      bfu ub;
      const __bf16* bp = &Bs[buf][(wn * 64 + t * 16 + l15) * LDS_STRIDE + hs * 8];
      ub.h[0] = *(const v8bf*)bp;
      ub.h[1] = *(const v8bf*)(bp + 16);
#pragma unroll
      for (int s = 0; s < 2; ++s) acc[s][t] = WMMA_BF16(af[s], ub.v, acc[s][t]);
    }
  };

  // double-buffered pipeline
  issue(0, 0);
  int cur = 0, k0 = 0;
  for (; k0 + 32 < K; k0 += 32) {
    issue(cur ^ 1, k0 + 32);  // prefetch next K-step (other buffer)
    WAIT_ASYNC(4);            // the 4 loads of the *current* buffer are done
    __syncthreads();
    compute(cur);
    __syncthreads();          // everyone done reading before buffer reuse
    cur ^= 1;
  }
  WAIT_ASYNC(0);
  __syncthreads();
  compute(cur);

#pragma unroll
  for (int s = 0; s < 2; ++s)
#pragma unroll
    for (int t = 0; t < 4; ++t)
#pragma unroll
      for (int r = 0; r < 8; ++r)
        C[(size_t)(mb + wm * 32 + s * 16 + r + 8 * hs) * N + nb + wn * 64 + t * 16 + l15] =
            acc[s][t][r];
}

// ---------------------------------------------------------------------------
// RMS norm per row: out(bf16) = in * rsqrt(mean(in^2) + eps) * w
// ---------------------------------------------------------------------------
__global__ __launch_bounds__(256) void rmsnorm_bf16(const float* __restrict__ x, int xs, int D,
                                                    const float* __restrict__ w,
                                                    __bf16* __restrict__ y, int ys) {
  const int row = blockIdx.x;
  const float* xr = x + (size_t)row * xs;
  float s = 0.f;
  for (int i = threadIdx.x; i < D; i += 256) { float v = xr[i]; s += v * v; }
  __shared__ float red[256];
  red[threadIdx.x] = s;
  __syncthreads();
  for (int off = 128; off > 0; off >>= 1) {
    if ((int)threadIdx.x < off) red[threadIdx.x] += red[threadIdx.x + off];
    __syncthreads();
  }
  const float inv = rsqrtf(red[0] / (float)D + 1e-6f);
  __bf16* yr = y + (size_t)row * ys;
  for (int i = threadIdx.x; i < D; i += 256) yr[i] = f2bf(xr[i] * inv * w[i]);
}

// ---------------------------------------------------------------------------
// RoPE + attention operand layout:
//   Qh [NH][S][192] bf16 : q_nope | roped q_pe
//   Kh [NH][S][192] bf16 : k_nope | roped shared k_pe
//   Vt [NH][128][S] bf16 : V transposed -> contiguous P@V B-fragments
// ---------------------------------------------------------------------------
__global__ __launch_bounds__(256) void prep_rope(const float* __restrict__ qf,
                                                 const float* __restrict__ kvf, int kvs,
                                                 const float* __restrict__ kvup,
                                                 __bf16* __restrict__ Qb,
                                                 __bf16* __restrict__ Kb,
                                                 __bf16* __restrict__ Vt, int S) {
  const int s = blockIdx.x;
  __shared__ float cs[32], sn[32], kpe[64];
  if (threadIdx.x < 32) {
    const int i = threadIdx.x;
    const float freq = __expf(-__logf(10000.f) * (2.f * (float)i) / 64.f);
    const float ang = (float)s * freq;
    sn[i] = __sinf(ang);
    cs[i] = __cosf(ang);
    const float x1 = kvf[(size_t)s * kvs + 512 + 2 * i];
    const float x2 = kvf[(size_t)s * kvs + 512 + 2 * i + 1];
    kpe[2 * i]     = x1 * cs[i] - x2 * sn[i];
    kpe[2 * i + 1] = x1 * sn[i] + x2 * cs[i];
  }
  __syncthreads();
  for (int idx = threadIdx.x; idx < 32 * 192; idx += 256) {
    const int h = idx / 192, d = idx % 192;
    const size_t o = ((size_t)h * S + s) * 192 + d;
    float qv;
    if (d < 128) {
      qv = qf[(size_t)s * 6144 + h * 192 + d];
    } else {
      const int j = d - 128, p = j >> 1;
      const float x1 = qf[(size_t)s * 6144 + h * 192 + 128 + 2 * p];
      const float x2 = qf[(size_t)s * 6144 + h * 192 + 128 + 2 * p + 1];
      qv = (j & 1) ? (x1 * sn[p] + x2 * cs[p]) : (x1 * cs[p] - x2 * sn[p]);
    }
    Qb[o] = f2bf(qv);
    const float kvv = (d < 128) ? kvup[(size_t)s * 8192 + h * 256 + d] : kpe[d - 128];
    Kb[o] = f2bf(kvv);
  }
  for (int idx = threadIdx.x; idx < 32 * 128; idx += 256) {
    const int h = idx / 128, d = idx % 128;
    Vt[((size_t)h * 128 + d) * S + s] = f2bf(kvup[(size_t)s * 8192 + h * 256 + 128 + d]);
  }
}

// ---------------------------------------------------------------------------
// Causal flash attention, all-WMMA. grid = (S/128, NH); each wave owns 16
// query rows, streams keys in chunks of 32, online softmax with ds_swizzle
// row reductions; P goes through LDS (s_wait_dscnt wave-local sync) to
// convert C-layout -> A-layout for the P@V WMMAs.
// ---------------------------------------------------------------------------
__global__ __launch_bounds__(256) void mla_attn(const __bf16* __restrict__ Qb,
                                                const __bf16* __restrict__ Kb,
                                                const __bf16* __restrict__ Vt,
                                                __bf16* __restrict__ attn, int S) {
  __shared__ __bf16 pshm[8][16 * 32];
  const int wave = threadIdx.x >> 5;
  const int lane = threadIdx.x & 31;
  const int l15 = lane & 15;
  const int hs  = lane >> 4;
  const int h = blockIdx.y;
  const int qbase = blockIdx.x * 128 + wave * 16;
  const __bf16* Qh = Qb + (size_t)h * S * 192;
  const __bf16* Kh = Kb + (size_t)h * S * 192;
  const __bf16* Vh = Vt + (size_t)h * 128 * S;
  __bf16* pt = &pshm[wave][0];

  v16bf qfr[6];
  {
    const int qrow = qbase + l15;
#pragma unroll
    for (int f = 0; f < 6; ++f) {
      bfu u;
      const __bf16* p = Qh + (size_t)qrow * 192 + f * 32 + hs * 16;
      u.h[0] = *(const v8bf*)p;
      u.h[1] = *(const v8bf*)(p + 8);
      qfr[f] = u.v;
    }
  }

  const v8f vzero = {0.f, 0.f, 0.f, 0.f, 0.f, 0.f, 0.f, 0.f};
  v8f o[8];
#pragma unroll
  for (int t = 0; t < 8; ++t) o[t] = vzero;
  float m_i[8], l_i[8];
#pragma unroll
  for (int r = 0; r < 8; ++r) { m_i[r] = -3.0e38f; l_i[r] = 0.f; }

  const float scale = 0.07216878364870322f;  // 1/sqrt(192)
  const int kend = qbase + 16;

  for (int kb = 0; kb < kend; kb += 32) {
    v8f s0 = vzero, s1 = vzero;
#pragma unroll
    for (int f = 0; f < 6; ++f) {
      const int k0 = f * 32 + hs * 16;
      const v16bf b0 = *(const v16bf*)(Kh + (size_t)(kb + l15) * 192 + k0);
      s0 = WMMA_BF16(qfr[f], b0, s0);
      const v16bf b1 = *(const v16bf*)(Kh + (size_t)(kb + 16 + l15) * 192 + k0);
      s1 = WMMA_BF16(qfr[f], b1, s1);
    }

    float p0[8], p1[8], alpha[8];
#pragma unroll
    for (int r = 0; r < 8; ++r) {
      const int mg = qbase + r + 8 * hs;
      const float v0 = (kb + l15      <= mg) ? s0[r] * scale : -3.0e38f;
      const float v1 = (kb + 16 + l15 <= mg) ? s1[r] * scale : -3.0e38f;
      float rm = half_max16(fmaxf(v0, v1));
      const float mn = fmaxf(m_i[r], rm);
      alpha[r] = __expf(m_i[r] - mn);
      m_i[r] = mn;
      p0[r] = __expf(v0 - mn);
      p1[r] = __expf(v1 - mn);
      const float rs = half_sum16(p0[r] + p1[r]);
      l_i[r] = l_i[r] * alpha[r] + rs;
    }
#pragma unroll
    for (int t = 0; t < 8; ++t)
#pragma unroll
      for (int r = 0; r < 8; ++r) o[t][r] *= alpha[r];

#pragma unroll
    for (int r = 0; r < 8; ++r) {
      const int row = r + 8 * hs;
      pt[row * 32 + l15]      = f2bf(p0[r]);
      pt[row * 32 + 16 + l15] = f2bf(p1[r]);
    }
    asm volatile("s_wait_dscnt 0x0" ::: "memory");  // wave-local LDS exchange
    v16bf pa;
    {
      bfu u;
      const __bf16* pp = pt + l15 * 32 + hs * 16;
      u.h[0] = *(const v8bf*)pp;
      u.h[1] = *(const v8bf*)(pp + 8);
      pa = u.v;
    }
#pragma unroll
    for (int t = 0; t < 8; ++t) {
      const v16bf vb = *(const v16bf*)(Vh + (size_t)(t * 16 + l15) * S + kb + hs * 16);
      o[t] = WMMA_BF16(pa, vb, o[t]);
    }
  }

  float inv[8];
#pragma unroll
  for (int r = 0; r < 8; ++r) inv[r] = (l_i[r] > 0.f) ? 1.f / l_i[r] : 0.f;
#pragma unroll
  for (int t = 0; t < 8; ++t)
#pragma unroll
    for (int r = 0; r < 8; ++r)
      attn[(size_t)(qbase + r + 8 * hs) * 4096 + (size_t)h * 128 + t * 16 + l15] =
          f2bf(o[t][r] * inv[r]);
}

// ---------------------------------------------------------------------------
// Host-side orchestration
// ---------------------------------------------------------------------------
extern "C" void kernel_launch(void* const* d_in, const int* in_sizes, int n_in,
                              void* d_out, int out_size, void* d_ws, size_t ws_size,
                              hipStream_t stream) {
  (void)in_sizes; (void)n_in; (void)out_size; (void)ws_size;
  constexpr int S = 2048, H = 4096, NH = 32, QL = 1536, KVL = 512;
  constexpr int QDIM = NH * 192;   // 6144
  constexpr int KVUP = NH * 256;   // 8192
  constexpr int KVA  = KVL + 64;   // 576 (valid cols)
  constexpr int KVAP = 640;        // padded to a multiple of 128 for the GEMM

  const float* x     = (const float*)d_in[0];
  const float* wq_a  = (const float*)d_in[1];
  const float* qnw   = (const float*)d_in[2];
  const float* wq_b  = (const float*)d_in[3];
  const float* wkv_a = (const float*)d_in[4];
  const float* kvnw  = (const float*)d_in[5];
  const float* wkv_b = (const float*)d_in[6];
  const float* wo    = (const float*)d_in[7];
  float* out = (float*)d_out;

  char* ws = (char*)d_ws;
  size_t off = 0;
  auto alloc = [&](size_t bytes) { char* p = ws + off; off += (bytes + 255) & ~(size_t)255; return p; };

  __bf16* xb     = (__bf16*)alloc((size_t)S * H * 2);
  __bf16* wqa_b  = (__bf16*)alloc((size_t)QL * H * 2);
  __bf16* wqb_b  = (__bf16*)alloc((size_t)QDIM * QL * 2);
  __bf16* wkva_b = (__bf16*)alloc((size_t)KVAP * H * 2);   // rows 576..639 zeroed
  __bf16* wkvb_b = (__bf16*)alloc((size_t)KVUP * KVL * 2);
  __bf16* wo_b   = (__bf16*)alloc((size_t)H * H * 2);
  float*  qa_f   = (float*)alloc((size_t)S * QL * 4);
  __bf16* qn_b   = (__bf16*)alloc((size_t)S * QL * 2);
  float*  q_f    = (float*)alloc((size_t)S * QDIM * 4);
  float*  kv_f   = (float*)alloc((size_t)S * KVAP * 4);
  __bf16* kvn_b  = (__bf16*)alloc((size_t)S * KVL * 2);
  float*  kvup_f = (float*)alloc((size_t)S * KVUP * 4);
  __bf16* Qb     = (__bf16*)alloc((size_t)NH * S * 192 * 2);
  __bf16* Kb     = (__bf16*)alloc((size_t)NH * S * 192 * 2);
  __bf16* Vt     = (__bf16*)alloc((size_t)NH * 128 * S * 2);
  __bf16* attn_b = (__bf16*)alloc((size_t)S * H * 2);

  // fp32 -> bf16 conversions (+ zero pad for wkv_a)
  cvt_f32_bf16<<<4096, 256, 0, stream>>>(x,     xb,     (long)S * H);
  cvt_f32_bf16<<<4096, 256, 0, stream>>>(wq_a,  wqa_b,  (long)QL * H);
  cvt_f32_bf16<<<4096, 256, 0, stream>>>(wq_b,  wqb_b,  (long)QDIM * QL);
  cvt_f32_bf16<<<4096, 256, 0, stream>>>(wkv_a, wkva_b, (long)KVA * H);
  zero_bf16<<<512, 256, 0, stream>>>(wkva_b + (size_t)KVA * H, (long)(KVAP - KVA) * H);
  cvt_f32_bf16<<<4096, 256, 0, stream>>>(wkv_b, wkvb_b, (long)KVUP * KVL);
  cvt_f32_bf16<<<4096, 256, 0, stream>>>(wo,    wo_b,   (long)H * H);

  // q_a = x @ wq_a.T ; rmsnorm ; q = qn @ wq_b.T
  gemm_bf16<<<dim3(QL / 128, S / 128), 256, 0, stream>>>(xb, wqa_b, qa_f, S, QL, H);
  rmsnorm_bf16<<<S, 256, 0, stream>>>(qa_f, QL, QL, qnw, qn_b, QL);
  gemm_bf16<<<dim3(QDIM / 128, S / 128), 256, 0, stream>>>(qn_b, wqb_b, q_f, S, QDIM, QL);

  // kv = x @ wkv_a.T (padded N) ; rmsnorm(kv_c) ; kv_up = kvn @ wkv_b.T
  gemm_bf16<<<dim3(KVAP / 128, S / 128), 256, 0, stream>>>(xb, wkva_b, kv_f, S, KVAP, H);
  rmsnorm_bf16<<<S, 256, 0, stream>>>(kv_f, KVAP, KVL, kvnw, kvn_b, KVL);
  gemm_bf16<<<dim3(KVUP / 128, S / 128), 256, 0, stream>>>(kvn_b, wkvb_b, kvup_f, S, KVUP, KVL);

  // RoPE + attention layout
  prep_rope<<<S, 256, 0, stream>>>(q_f, kv_f, KVAP, kvup_f, Qb, Kb, Vt, S);

  // causal flash attention
  mla_attn<<<dim3(S / 128, NH), 256, 0, stream>>>(Qb, Kb, Vt, attn_b, S);

  // out = attn @ wo.T
  gemm_bf16<<<dim3(H / 128, S / 128), 256, 0, stream>>>(attn_b, wo_b, out, S, H, H);
}
